// QKV_27161373180220
// MI455X (gfx1250) — compile-verified, matrix operand order checked by reading, not run on hardware
//
#include <hip/hip_runtime.h>
#include <hip/hip_fp16.h>

// ---------------------------------------------------------------------------
// MI455X (gfx1250) conv-attention. Heavy convs -> implicit-GEMM with
// v_wmma_f32_16x16x32_f16 (f32 accumulate). wave32, 8 waves/block,
// 128x128x32 block tile, 32x64 wave tile, double-buffered LDS,
// branchless (clamp+select) staging, vectorized global/LDS traffic.
// ---------------------------------------------------------------------------

typedef __attribute__((ext_vector_type(16))) _Float16 v16h;
typedef __attribute__((ext_vector_type(8)))  _Float16 v8h;
typedef __attribute__((ext_vector_type(8)))  float    v8f;
typedef __attribute__((ext_vector_type(4)))  float    v4f;

#define BM 128
#define BN 128
#define BK 32

// epilogue modes
#define EPI_BN_LRELU   1   // y = lrelu(bn(conv+bias))
#define EPI_LRELU_BN   2   // y = bn(lrelu(conv+bias))      (p3)
#define EPI_TANH       3   // y = tanh(conv+bias)           (k3, p4)
#define EPI_BN_ONLY    4   // y = bn(conv+bias)             (d2)

__device__ __forceinline__ int imin(int a, int b) { return a < b ? a : b; }
__device__ __forceinline__ int imax(int a, int b) { return a > b ? a : b; }

// ---------------------------------------------------------------------------
// Generic implicit-GEMM conv:  out[n,co,p,q] = act(bn(sum_{c,r,s} in*w + b))
//   GEMM view: D[m, j],  m = Cout,  j = (n,p,q),  k = (c,r,s)
// ---------------------------------------------------------------------------
__global__ __launch_bounds__(256)
void conv_gemm_wmma_kernel(const float* __restrict__ in,
                           const float* __restrict__ wgt,
                           const float* __restrict__ bias,
                           const float* __restrict__ bng,
                           const float* __restrict__ bnb,
                           float* __restrict__ out,
                           int N, int Cin, int H, int W,
                           int Cout, int R, int S, int stride, int pad,
                           int P, int Q, int mode)
{
    const int Ngemm = N * P * Q;
    const int Ktot  = Cin * R * S;
    const int PQ    = P * Q;
    const int RS    = R * S;
    const int HW    = H * W;

    __shared__ _Float16 As[2][BM * BK];   // [m][k] rows of 64B
    __shared__ _Float16 Bs[2][BN * BK];   // [j][k] rows of 64B

    const int tid  = threadIdx.x;
    const int wave = tid >> 5;
    const int lane = tid & 31;
    const int m0   = blockIdx.y * BM;
    const int j0   = blockIdx.x * BN;

    const int wm = wave & 3;              // 4 waves along M (32 rows each)
    const int wn = wave >> 2;             // 2 waves along N (64 cols each)

    // ---- per-thread staging geometry (branchless; computed once) ----
    // A: two threads per weight row, 16 contiguous k each
    const int  mrowA = tid >> 1;
    const int  kcolA = (tid & 1) * 16;
    const bool aRow  = (m0 + mrowA) < Cout;
    const long aBase = (long)(aRow ? (m0 + mrowA) : 0) * Ktot;   // clamped row
    const bool k4    = (Ktot & 3) == 0;

    // B: two threads per output column j, 16 contiguous k each
    const int  jjB   = tid & 127;
    const int  kcolB = (tid >> 7) * 16;
    const int  gjB   = j0 + jjB;
    const bool jok   = gjB < Ngemm;
    int nB, pB, qB;
    {
        const int gj = jok ? gjB : 0;
        nB = gj / PQ;
        const int pq = gj - nB * PQ;
        pB = pq / Q;
        qB = pq - pB * Q;
    }
    const int  hB    = pB * stride - pad;
    const int  wB    = qB * stride - pad;
    const long nOff  = (long)nB * Cin * HW;

    // ---- staging lambdas (unconditional clamped loads + vector LDS stores) --
    auto stageA = [&](int buf, int k0) {
        _Float16* dst = &As[buf][mrowA * BK + kcolA];
        const int gk0 = k0 + kcolA;
        v8h h0, h1;
        if (aRow && k4 && (gk0 + 16 <= Ktot)) {
            const v4f* src = (const v4f*)&wgt[aBase + gk0];
            const v4f f0 = src[0], f1 = src[1], f2 = src[2], f3 = src[3];
            #pragma unroll
            for (int e = 0; e < 4; e++) {
                h0[e]     = (_Float16)f0[e];
                h0[4 + e] = (_Float16)f1[e];
                h1[e]     = (_Float16)f2[e];
                h1[4 + e] = (_Float16)f3[e];
            }
        } else {
            #pragma unroll
            for (int i = 0; i < 16; i++) {
                const int  gk = gk0 + i;
                float v = wgt[aBase + imin(gk, Ktot - 1)];
                v = (aRow && gk < Ktot) ? v : 0.0f;
                if (i < 8) h0[i] = (_Float16)v; else h1[i - 8] = (_Float16)v;
            }
        }
        *(v8h*)dst       = h0;
        *(v8h*)(dst + 8) = h1;
    };

    auto stageB = [&](int buf, int k0) {
        const int gk0 = k0 + kcolB;
        int c, r, s;
        {   // decompose once (clamped); then incremental carry updates
            const int gk = imin(gk0, Ktot - 1);
            c = gk / RS;
            const int rs = gk - c * RS;
            r = rs / S;
            s = rs - r * S;
        }
        v8h h0, h1;
        #pragma unroll
        for (int i = 0; i < 16; i++) {
            const int gk = gk0 + i;
            const int hh = hB + r;
            const int ww = wB + s;
            const bool valid = jok && (gk < Ktot) &&
                               (hh >= 0) && (hh < H) && (ww >= 0) && (ww < W);
            const int hc = imin(imax(hh, 0), H - 1);
            const int wc = imin(imax(ww, 0), W - 1);
            const int cc = imin(c, Cin - 1);
            float v = in[nOff + (long)cc * HW + hc * W + wc];
            v = valid ? v : 0.0f;
            if (i < 8) h0[i] = (_Float16)v; else h1[i - 8] = (_Float16)v;
            s++;
            if (s == S) { s = 0; r++; if (r == R) { r = 0; c++; } }
        }
        _Float16* dst = &Bs[buf][jjB * BK + kcolB];
        *(v8h*)dst       = h0;
        *(v8h*)(dst + 8) = h1;
    };

    // ---- accumulators ----
    v8f acc[2][4];
    #pragma unroll
    for (int mi = 0; mi < 2; mi++)
        #pragma unroll
        for (int ni = 0; ni < 4; ni++)
            #pragma unroll
            for (int e = 0; e < 8; e++) acc[mi][ni][e] = 0.0f;

    const int lrow  = lane & 15;          // fragment row/col within 16
    const int khalf = (lane >> 4) * 8;    // ISA 16-bit A/B fragment layout

    // ---- main K loop, double buffered ----
    stageA(0, 0);
    stageB(0, 0);
    int pbuf = 0;
    for (int k0 = 0; k0 < Ktot; k0 += BK) {
        __syncthreads();
        if (k0 + BK < Ktot) {
            stageA(pbuf ^ 1, k0 + BK);
            stageB(pbuf ^ 1, k0 + BK);
            if (aRow && k0 + 2 * BK < Ktot)       // CDNA5 global_prefetch_b8
                __builtin_prefetch(&wgt[aBase + k0 + 2 * BK + kcolA], 0, 1);
        }
        // fragments: two ds_read_b128 each
        v16h afrag[2], bfrag[4];
        #pragma unroll
        for (int mi = 0; mi < 2; mi++) {
            const int mrow = wm * 32 + mi * 16 + lrow;
            const v8h lo = *(const v8h*)&As[pbuf][mrow * BK + khalf];
            const v8h hi = *(const v8h*)&As[pbuf][mrow * BK + 16 + khalf];
            #pragma unroll
            for (int e = 0; e < 8; e++) { afrag[mi][e] = lo[e]; afrag[mi][8 + e] = hi[e]; }
        }
        #pragma unroll
        for (int ni = 0; ni < 4; ni++) {
            const int jcol = wn * 64 + ni * 16 + lrow;
            const v8h lo = *(const v8h*)&Bs[pbuf][jcol * BK + khalf];
            const v8h hi = *(const v8h*)&Bs[pbuf][jcol * BK + 16 + khalf];
            #pragma unroll
            for (int e = 0; e < 8; e++) { bfrag[ni][e] = lo[e]; bfrag[ni][8 + e] = hi[e]; }
        }
        #pragma unroll
        for (int mi = 0; mi < 2; mi++)
            #pragma unroll
            for (int ni = 0; ni < 4; ni++)
                acc[mi][ni] = __builtin_amdgcn_wmma_f32_16x16x32_f16(
                    false, afrag[mi], false, bfrag[ni],
                    (short)0, acc[mi][ni], false, false);
        pbuf ^= 1;
    }

    // ---- fused epilogue: bias + BN(eval) + activation, NCHW scatter ----
    const float inv_sqrt = rsqrtf(1.0f + 1e-5f);
    // wave-uniform full-tile flag, forced into an SGPR branch
    const int full = __builtin_amdgcn_readfirstlane(
        ((m0 + wm * 32 + 32 <= Cout) && (j0 + wn * 64 + 64 <= Ngemm)) ? 1 : 0);

    #pragma unroll
    for (int mi = 0; mi < 2; mi++) {
        #pragma unroll
        for (int ni = 0; ni < 4; ni++) {
            // j is fixed per (ni, lane): decompose once
            const int j  = j0 + wn * 64 + ni * 16 + (lane & 15);
            const int js = imin(j, Ngemm - 1);
            const int n  = js / PQ;
            const int pq = js - n * PQ;
            #pragma unroll
            for (int rr = 0; rr < 8; rr++) {
                // ISA C/D layout: VGPR rr -> M=rr (lanes 0-15), M=rr+8 (lanes 16-31)
                const int m = m0 + wm * 32 + mi * 16 + rr + ((lane >= 16) ? 8 : 0);
                if (full || (m < Cout && j < Ngemm)) {
                    float y = acc[mi][ni][rr] + bias[m];
                    if (mode == EPI_BN_LRELU || mode == EPI_BN_ONLY) {
                        y = y * (bng[m] * inv_sqrt) + bnb[m];
                        if (mode == EPI_BN_LRELU) y = (y > 0.0f) ? y : 0.2f * y;
                    } else if (mode == EPI_LRELU_BN) {
                        y = (y > 0.0f) ? y : 0.2f * y;
                        y = y * (bng[m] * inv_sqrt) + bnb[m];
                    } else if (mode == EPI_TANH) {
                        y = tanhf(y);
                    }
                    out[((long)n * Cout + m) * PQ + pq] = y;
                }
            }
        }
    }
}

// ---------------------------------------------------------------------------
// Small scalar kernels (together <2% of FLOPs)
// ---------------------------------------------------------------------------

// out[o] = tanh(mean(in[o, 0:hw]))
__global__ void pool_tanh_kernel(const float* __restrict__ in,
                                 float* __restrict__ out, int nout, int hw)
{
    const int o = blockIdx.x * blockDim.x + threadIdx.x;
    if (o >= nout) return;
    const float* p = in + (long)o * hw;
    float s = 0.0f;
    for (int i = 0; i < hw; i++) s += p[i];
    out[o] = tanhf(s / (float)hw);
}

// ragged gather: q_pad[b,m,c] = c_pool[clip(off_b+m)] * (m < c_num[b])
__global__ void qpad_kernel(const float* __restrict__ cpool,
                            const int* __restrict__ cnum,
                            float* __restrict__ qpad,
                            int bs, int maxc, int C, int ntot)
{
    const int idx = blockIdx.x * blockDim.x + threadIdx.x;
    const int total = bs * maxc * C;
    if (idx >= total) return;
    const int ch = idx % C;
    const int t  = idx / C;
    const int m  = t % maxc;
    const int b  = t / maxc;
    int off = 0;
    for (int i = 0; i < b; i++) off += cnum[i];
    int src = off + m;
    if (src < 0) src = 0;
    if (src > ntot - 1) src = ntot - 1;
    qpad[idx] = (m < cnum[b]) ? cpool[(long)src * C + ch] : 0.0f;
}

// per (b,f) block: qk = [q_pad;qx] @ k, then qkv rows 0..10 and xsim row 11
__global__ __launch_bounds__(256)
void attn_qkv_kernel(const float* __restrict__ k,     // [64, 2048, 84]
                     const float* __restrict__ v,     // [64, 256, 84]
                     const float* __restrict__ qpad,  // [8, 11, 2048]
                     const float* __restrict__ qx,    // [64, 2048] == [8,8,2048]
                     float* __restrict__ cpred)       // [64, 13, 256]
{
    const int L = 84, CK = 2048, CV = 256, MAXC = 11, FS = 8, ROWS = 19;
    const int bf = blockIdx.x;
    const int b  = bf / FS;
    const float* kbase = k + (long)bf * CK * L;
    const float* vbase = v + (long)bf * CV * L;

    __shared__ float qk[ROWS * L];
    __shared__ float kt[32 * L];
    __shared__ float qt[ROWS * 32];
    __shared__ float xbar[L];
    const int tid = threadIdx.x;

    for (int i = tid; i < ROWS * L; i += 256) qk[i] = 0.0f;
    __syncthreads();

    for (int c0 = 0; c0 < CK; c0 += 32) {
        for (int i = tid; i < 32 * L; i += 256)
            kt[i] = kbase[(long)(c0 + i / L) * L + (i % L)];
        for (int i = tid; i < ROWS * 32; i += 256) {
            const int rr = i / 32, cc = i % 32;
            qt[i] = (rr < MAXC)
                  ? qpad[((long)b * MAXC + rr) * CK + c0 + cc]
                  : qx[((long)b * FS + (rr - MAXC)) * CK + c0 + cc];
        }
        __syncthreads();
        for (int i = tid; i < ROWS * L; i += 256) {
            const int rr = i / L, l = i % L;
            float s = qk[i];
            #pragma unroll 8
            for (int cc = 0; cc < 32; cc++) s += qt[rr * 32 + cc] * kt[cc * L + l];
            qk[i] = s;
        }
        __syncthreads();
    }

    for (int l = tid; l < L; l += 256) {
        float s = 0.0f;
        for (int g = 0; g < FS; g++) s += qk[(MAXC + g) * L + l];
        xbar[l] = s * (1.0f / (float)FS);
    }
    __syncthreads();

    for (int i = tid; i < 12 * CV; i += 256) {
        const int rr = i / CV, vc = i % CV;
        const float* qrow = (rr < MAXC) ? &qk[rr * L] : xbar;
        const float* vrow = vbase + (long)vc * L;
        float s = 0.0f;
        for (int l = 0; l < L; l++) s += qrow[l] * vrow[l];
        cpred[((long)bf * 13 + rr) * CV + vc] = s;
    }
}

// out[o] = mean(in[o, 0:L])
__global__ void mean_l_kernel(const float* __restrict__ in,
                              float* __restrict__ out, int nrows, int L)
{
    const int o = blockIdx.x * blockDim.x + threadIdx.x;
    if (o >= nrows) return;
    const float* p = in + (long)o * L;
    float s = 0.0f;
    for (int l = 0; l < L; l++) s += p[l];
    out[o] = s / (float)L;
}

// per-b block: kk = km km^T ; tkkv = kk @ vm -> cpred row 12
__global__ __launch_bounds__(256)
void tkkv_kernel(const float* __restrict__ km,   // [64, 2048]
                 const float* __restrict__ vm,   // [64, 256]
                 float* __restrict__ cpred)      // [64, 13, 256]
{
    const int FS = 8, CK = 2048, CV = 256;
    const int b = blockIdx.x;
    __shared__ float kk[FS * FS];
    const int tid = threadIdx.x;
    if (tid < FS * FS) {
        const int f = tid / FS, g = tid % FS;
        const float* a = km + ((long)b * FS + f) * CK;
        const float* c = km + ((long)b * FS + g) * CK;
        float s = 0.0f;
        for (int i = 0; i < CK; i++) s += a[i] * c[i];
        kk[tid] = s;
    }
    __syncthreads();
    for (int i = tid; i < FS * CV; i += 256) {
        const int f = i / CV, vc = i % CV;
        float s = 0.0f;
        for (int g = 0; g < FS; g++) s += kk[f * FS + g] * vm[((long)b * FS + g) * CV + vc];
        cpred[(((long)b * FS + f) * 13 + 12) * CV + vc] = s;
    }
}

// ---------------------------------------------------------------------------
// Host-side orchestration
// ---------------------------------------------------------------------------
static void launch_conv(hipStream_t st, const float* in, const float* w,
                        const float* b, const float* g, const float* be,
                        float* out, int N, int Cin, int H, int W,
                        int Cout, int R, int S, int stride, int pad, int mode)
{
    const int P = (H + 2 * pad - R) / stride + 1;
    const int Q = (W + 2 * pad - S) / stride + 1;
    const int Ng = N * P * Q;
    dim3 grid((Ng + BN - 1) / BN, (Cout + BM - 1) / BM);
    conv_gemm_wmma_kernel<<<grid, dim3(256), 0, st>>>(
        in, w, b, g, be, out, N, Cin, H, W, Cout, R, S, stride, pad, P, Q, mode);
}

extern "C" void kernel_launch(void* const* d_in, const int* in_sizes, int n_in,
                              void* d_out, int out_size, void* d_ws, size_t ws_size,
                              hipStream_t stream)
{
    (void)in_sizes; (void)n_in; (void)out_size; (void)ws_size;
    const int BS = 8, FS = 8, MAXC = 11, NTOT = BS * MAXC, NBF = BS * FS;

    const float* c    = (const float*)d_in[0];
    const float* x    = (const float*)d_in[1];
    const int*   cnum = (const int*)  d_in[2];
    // d_in[3]=bs, d_in[4]=fs: compile-time constants (8,8)
    int pi = 5;
    auto P = [&](int i) { return (const float*)d_in[pi + i]; };
    const float *k1w=P(0),*k1b=P(1),*k1g=P(2),*k1e=P(3);
    const float *k2w=P(4),*k2b=P(5),*k2g=P(6),*k2e=P(7);
    const float *k3w=P(8),*k3b=P(9);
    const float *v1w=P(10),*v1b=P(11),*v1g=P(12),*v1e=P(13);
    const float *v2w=P(14),*v2b=P(15),*v2g=P(16),*v2e=P(17);
    const float *v3w=P(18),*v3b=P(19),*v3g=P(20),*v3e=P(21);
    const float *v4w=P(22),*v4b=P(23),*v4g=P(24),*v4e=P(25);
    const float *d1w=P(26),*d1b=P(27),*d1g=P(28),*d1e=P(29);
    const float *d2w=P(30),*d2b=P(31),*d2g=P(32),*d2e=P(33);
    const float *p1w=P(34),*p1b=P(35),*p1g=P(36),*p1e=P(37);
    const float *p2w=P(38),*p2b=P(39),*p2g=P(40),*p2e=P(41);
    const float *p3w=P(42),*p3b=P(43),*p3g=P(44),*p3e=P(45);
    const float *p4w=P(46),*p4b=P(47);

    float* ws  = (float*)d_ws;
    float* fo  = (float*)d_out;                 // [704 pred][32768 kqkv]
    // workspace layout (floats); dead regions reused by the pred stage
    float* cpool = ws + 0;                      // 180224
    float* qpad  = ws + 180224;                 // 180224
    float* qx    = ws + 360448;                 // 131072
    float* bufA  = ws + 491520;                 // 5505024  (k1/v1 out; later p1h)
    float* bufB  = ws + 5996544;                // 2752512  (k2/v2 out)
    float* kbuf  = ws + 8749056;                // 11010048 (k; later d1h)
    float* hv3   = ws + 19759104;               // 2752512  (v3 out; later p2h)
    float* vbuf  = ws + 22511616;               // 1376256  (v; later p3h)
    float* cpred = ws + 23887872;               // 212992
    float* km    = ws + 24100864;               // 131072
    float* vm    = ws + 24231936;               // 16384
    float* d1h   = kbuf;                        // reuse (k dead after attention)
    float* p1h   = bufA;                        // reuse (spans bufA..kbuf, all dead)
    float* p2h   = hv3;
    float* p3h   = vbuf;
    float* kqkv  = fo + 704;                    // second output, also p1 input

    // ---- queries ----
    pool_tanh_kernel<<<(NTOT*2048+255)/256, 256, 0, stream>>>(c, cpool, NTOT*2048, 128);
    pool_tanh_kernel<<<(NBF *2048+255)/256, 256, 0, stream>>>(x, qx,    NBF *2048, 128);
    qpad_kernel<<<(BS*MAXC*2048+255)/256, 256, 0, stream>>>(cpool, cnum, qpad, BS, MAXC, 2048, NTOT);

    // ---- key branch (WMMA implicit-GEMM convs) ----
    launch_conv(stream, x,    k1w, k1b, k1g, k1e, bufA, NBF, 2048, 16, 8, 1024, 3, 3, 1, 0, EPI_BN_LRELU);
    launch_conv(stream, bufA, k2w, k2b, k2g, k2e, bufB, NBF, 1024, 14, 6,  512, 1, 1, 1, 0, EPI_BN_LRELU);
    launch_conv(stream, bufB, k3w, k3b, 0,   0,   kbuf, NBF,  512, 14, 6, 2048, 1, 1, 1, 0, EPI_TANH);

    // ---- value branch ----
    launch_conv(stream, x,    v1w, v1b, v1g, v1e, bufA, NBF, 2048, 16, 8, 1024, 3, 3, 1, 0, EPI_BN_LRELU);
    launch_conv(stream, bufA, v2w, v2b, v2g, v2e, bufB, NBF, 1024, 14, 6,  512, 1, 1, 1, 0, EPI_BN_LRELU);
    launch_conv(stream, bufB, v3w, v3b, v3g, v3e, hv3,  NBF,  512, 14, 6,  512, 1, 1, 1, 0, EPI_BN_LRELU);
    launch_conv(stream, hv3,  v4w, v4b, v4g, v4e, vbuf, NBF,  512, 14, 6,  256, 1, 1, 1, 0, EPI_BN_LRELU);

    // ---- attention: qkv / xsim / tkkv -> cpred [64,13,16,16] ----
    attn_qkv_kernel<<<NBF, 256, 0, stream>>>(kbuf, vbuf, qpad, qx, cpred);
    mean_l_kernel<<<(NBF*2048+255)/256, 256, 0, stream>>>(kbuf, km, NBF*2048, 84);
    mean_l_kernel<<<(NBF* 256+255)/256, 256, 0, stream>>>(vbuf, vm, NBF* 256, 84);
    tkkv_kernel<<<BS, 256, 0, stream>>>(km, vm, cpred);

    // ---- down: kqkv2class ----
    launch_conv(stream, cpred, d1w, d1b, d1g, d1e, d1h,  NBF,  13, 16, 16,  512, 1, 1, 1, 0, EPI_BN_LRELU);
    launch_conv(stream, d1h,   d2w, d2b, d2g, d2e, kqkv, NBF, 512, 16, 16,    2, 3, 3, 1, 1, EPI_BN_ONLY);

    // ---- pred: pred_bing ----
    launch_conv(stream, kqkv, p1w, p1b, p1g, p1e, p1h, NBF,    2, 16, 16, 1024, 3, 3, 1, 0, EPI_BN_LRELU);
    launch_conv(stream, p1h,  p2w, p2b, p2g, p2e, p2h, NBF, 1024, 14, 14,  512, 3, 3, 2, 0, EPI_BN_LRELU);
    launch_conv(stream, p2h,  p3w, p3b, p3g, p3e, p3h, NBF,  512,  6,  6,  256, 3, 3, 1, 0, EPI_LRELU_BN);
    launch_conv(stream, p3h,  p4w, p4b, 0,   0,   fo,  NBF,  256,  4,  4, MAXC, 4, 4, 1, 0, EPI_TANH);
}
